// TransformerBlock_70497593196684
// MI455X (gfx1250) — matrix operand order repaired and checked by reading, not compiled, not run on hardware
//
#include <hip/hip_runtime.h>

// ---------------------------------------------------------------------------
// Types for CDNA5 WMMA (wave32): v_wmma_f32_16x16x32_bf16
// ---------------------------------------------------------------------------
typedef __attribute__((ext_vector_type(16))) __bf16 v16bf;
typedef __attribute__((ext_vector_type(8)))  float  v8f;
typedef __attribute__((ext_vector_type(4)))  unsigned int v4u32;
typedef __attribute__((ext_vector_type(8)))  int v8i32;
typedef __attribute__((ext_vector_type(4)))  int v4i32;

union Frag { v16bf v; uint4 q[2]; };          // 16 bf16 = 32 bytes
union AccF { v8f  v; float f[8]; };           // 16x16 f32 C/D tile per lane

__device__ inline v8f wmma_bf16(const Frag& a, const Frag& b, v8f c) {
  return __builtin_amdgcn_wmma_f32_16x16x32_bf16(
      /*neg_a=*/false, a.v, /*neg_b=*/false, b.v,
      /*c_mod=*/(short)0, c, /*reuse_a=*/false, /*reuse_b=*/false);
}

__device__ inline unsigned short f2b(float f) {
  union { float f; unsigned u; } x; x.f = f;
  unsigned u = x.u;
  unsigned r = (u + 0x7FFFu + ((u >> 16) & 1u)) >> 16;   // RNE
  return (unsigned short)r;
}

__device__ inline float gelu_f(float x) {
  const float c = 0.7978845608028654f;                    // sqrt(2/pi)
  float t = tanhf(c * (x + 0.044715f * x * x * x));
  return 0.5f * x * (1.0f + t);
}

// ---------------------------------------------------------------------------
// Tensor Data Mover: 2D tile load (64 rows x 64 bf16 elems) into LDS.
// D# packed per CDNA5 ISA Group0/Group1 bit layout:
//   g0: [1:0]=count=1, [63:32]=lds_addr, [120:64]=global_addr, [127:126]=type=2
//   g1: [17:16]=data_size(2B), [79:48]=tensor_dim0, [111:80]=tensor_dim1,
//       [127:112]=tile_dim0, [143:128]=tile_dim1, [207:160]=tensor_dim0_stride
// Toolchain here = clang-23 6-arg builtin:
//   (uint32x4 g0, int32x8 g1, int32x4, int32x4, int32x8, i32 cpol)
// ---------------------------------------------------------------------------
__device__ inline void tdm_load_tile64x64(const void* gsrc, unsigned lds_off,
                                          unsigned row_stride_elems) {
  unsigned long long ga = (unsigned long long)(uintptr_t)gsrc;
  v4u32 g0;
  g0[0] = 1u;                                             // count = 1
  g0[1] = lds_off;                                        // LDS byte address
  g0[2] = (unsigned)ga;                                   // global addr lo
  g0[3] = (unsigned)((ga >> 32) & 0x01FFFFFFull) | (2u << 30);  // hi + type=2
  unsigned td0 = row_stride_elems;                        // tensor dim0 (elems)
  unsigned td1 = 1u << 20;                                // tensor dim1 (big)
  v8i32 g1;
  g1[0] = (int)(1u << 16);                                // data_size = 2 bytes
  g1[1] = (int)((td0 & 0xFFFFu) << 16);                   // tensor_dim0[15:0]
  g1[2] = (int)(((td0 >> 16) & 0xFFFFu) | ((td1 & 0xFFFFu) << 16));
  g1[3] = (int)(((td1 >> 16) & 0xFFFFu) | (64u << 16));   // | tile_dim0 = 64
  g1[4] = (int)64;                                        // tile_dim1 = 64
  g1[5] = (int)td0;                                       // dim0_stride lo32
  g1[6] = 0;
  g1[7] = 0;
  v4i32 z4 = {0, 0, 0, 0};
  v8i32 z8 = {0, 0, 0, 0, 0, 0, 0, 0};
  __builtin_amdgcn_tensor_load_to_lds(g0, g1, z4, z4, z8, 0);
}

// ---------------------------------------------------------------------------
// fp32 -> bf16 conversion (weights)
// ---------------------------------------------------------------------------
__global__ void cvt_bf16_kernel(const float* __restrict__ in,
                                unsigned short* __restrict__ out, int n) {
  int i = blockIdx.x * blockDim.x + threadIdx.x;
  int stride = gridDim.x * blockDim.x;
  for (; i < n; i += stride) out[i] = f2b(in[i]);
}

// ---------------------------------------------------------------------------
// LayerNorm over D=1024, one block (256 thr) per row, bf16 output
// ---------------------------------------------------------------------------
__global__ __launch_bounds__(256)
void ln_bf16_kernel(const float* __restrict__ x, const float* __restrict__ g,
                    const float* __restrict__ be, unsigned short* __restrict__ out,
                    int Dd) {
  int row = blockIdx.x;
  int tid = threadIdx.x;
  const float* xr = x + (size_t)row * Dd;
  float4 v = ((const float4*)xr)[tid];
  float s  = v.x + v.y + v.z + v.w;
  float s2 = v.x * v.x + v.y * v.y + v.z * v.z + v.w * v.w;
  __shared__ float red[2][8];
  for (int m = 16; m > 0; m >>= 1) {
    s  += __shfl_xor(s,  m, 32);
    s2 += __shfl_xor(s2, m, 32);
  }
  if ((tid & 31) == 0) { red[0][tid >> 5] = s; red[1][tid >> 5] = s2; }
  __syncthreads();
  s = 0.f; s2 = 0.f;
  for (int i = 0; i < 8; i++) { s += red[0][i]; s2 += red[1][i]; }
  float mean = s / Dd;
  float var  = s2 / Dd - mean * mean;
  float rstd = rsqrtf(var + 1e-5f);
  int base = tid * 4;
  unsigned short* orow = out + (size_t)row * Dd;
  float e[4] = {v.x, v.y, v.z, v.w};
  for (int i = 0; i < 4; i++)
    orow[base + i] = f2b((e[i] - mean) * rstd * g[base + i] + be[base + i]);
}

// ---------------------------------------------------------------------------
// bf16 WMMA GEMM: C[M,N] = A[M,K] @ B[K,N] (+bias, opt GELU, opt residual)
// Block tile 64x128, 256 threads = 8 waves, each wave a 16x64 C strip
// (one A fragment reused across 4 B fragments -> 4 WMMAs / K-step).
// Software pipelined: global loads for step k+1 issued before the WMMAs of
// step k, stored to LDS after them.
// ---------------------------------------------------------------------------
__global__ __launch_bounds__(256)
void gemm_bf16_kernel(const unsigned short* __restrict__ A,
                      const unsigned short* __restrict__ Bw,
                      const float* __restrict__ bias,
                      const float* __restrict__ resid,
                      float* __restrict__ outF,
                      unsigned short* __restrict__ outB,
                      int M, int N, int K, int doGelu) {
  __shared__ unsigned short lA[64 * 32];      // [row][k]
  __shared__ unsigned short lBt[128 * 32];    // transposed: [n][k]
  int tid = threadIdx.x;
  int lane = tid & 31, wv = tid >> 5;
  int wm = wv & 3, wn = wv >> 2;
  int rowbase = blockIdx.y * 64, colbase = blockIdx.x * 128;
  int m0 = wm * 16, n0 = wn * 64;

  AccF acc[4];
#pragma unroll
  for (int t = 0; t < 4; t++)
#pragma unroll
    for (int i = 0; i < 8; i++) acc[t].f[i] = 0.f;

  // staging assignments
  int mload  = tid >> 2;        // A: 0..63
  int kload  = (tid & 3) * 8;   // A: 0,8,16,24
  int kB     = tid >> 3;        // B: k row 0..31
  int nB0    = (tid & 7) * 8;   // B: col 0..56 (two halves: +0, +64)

  int mA  = lane & 15, kb8 = (lane >> 4) * 8;
  int nBl = lane & 15, kkb = (lane >> 4) * 8;

  int nsteps = K >> 5;

  // prologue: load step 0 into registers, stage to LDS
  uint4 pa, pb0, pb1;
  pa  = *(const uint4*)&A[(size_t)(rowbase + mload) * K + kload];
  pb0 = *(const uint4*)&Bw[(size_t)kB * N + colbase + nB0];
  pb1 = *(const uint4*)&Bw[(size_t)kB * N + colbase + 64 + nB0];
  {
    *(uint4*)&lA[mload * 32 + kload] = pa;
    union { uint4 q; unsigned short s[8]; } t0, t1;
    t0.q = pb0; t1.q = pb1;
#pragma unroll
    for (int i = 0; i < 8; i++) {
      lBt[(nB0 + i) * 32 + kB]      = t0.s[i];
      lBt[(64 + nB0 + i) * 32 + kB] = t1.s[i];
    }
  }
  __syncthreads();

  for (int kt = 0; kt < nsteps; kt++) {
    // prefetch next K-step while WMMAs execute
    if (kt + 1 < nsteps) {
      int kc = (kt + 1) * 32;
      pa  = *(const uint4*)&A[(size_t)(rowbase + mload) * K + kc + kload];
      pb0 = *(const uint4*)&Bw[(size_t)(kc + kB) * N + colbase + nB0];
      pb1 = *(const uint4*)&Bw[(size_t)(kc + kB) * N + colbase + 64 + nB0];
    }

    Frag af;
    af.q[0] = *(const uint4*)&lA[(m0 + mA) * 32 + kb8];
    af.q[1] = *(const uint4*)&lA[(m0 + mA) * 32 + 16 + kb8];
#pragma unroll
    for (int t = 0; t < 4; t++) {
      Frag bf;
      int nrow = (n0 + t * 16 + nBl) * 32;
      bf.q[0] = *(const uint4*)&lBt[nrow + kkb];
      bf.q[1] = *(const uint4*)&lBt[nrow + kkb + 16];
      acc[t].v = wmma_bf16(af, bf, acc[t].v);
    }
    __syncthreads();

    if (kt + 1 < nsteps) {
      *(uint4*)&lA[mload * 32 + kload] = pa;
      union { uint4 q; unsigned short s[8]; } t0, t1;
      t0.q = pb0; t1.q = pb1;
#pragma unroll
      for (int i = 0; i < 8; i++) {
        lBt[(nB0 + i) * 32 + kB]      = t0.s[i];
        lBt[(64 + nB0 + i) * 32 + kB] = t1.s[i];
      }
    }
    __syncthreads();
  }

  // epilogue: lane holds C(m = r + (lane>=16)*8, n = lane&15)
  int mh = (lane >> 4) * 8;
#pragma unroll
  for (int r = 0; r < 8; r++) {
    int row = rowbase + m0 + mh + r;
#pragma unroll
    for (int t = 0; t < 4; t++) {
      int col = colbase + n0 + t * 16 + nBl;
      float v = acc[t].f[r] + bias[col];
      if (doGelu) v = gelu_f(v);
      size_t o = (size_t)row * N + col;
      if (resid) v += resid[o];
      if (outF) outF[o] = v;
      if (outB) outB[o] = f2b(v);
    }
  }
}

// ---------------------------------------------------------------------------
// Flash attention (causal), head_dim=64. grid=(B*H, S/64), block=128 (4 waves).
// K tiles DMA'd by the Tensor Data Mover (double-buffered, issued by wave 0,
// synchronized with s_wait_tensorcnt + barrier); V staged transposed by the
// whole block while the K DMA is in flight.
// ---------------------------------------------------------------------------
__global__ __launch_bounds__(128)
void attn_kernel(const unsigned short* __restrict__ Q,
                 const unsigned short* __restrict__ Kb,
                 const unsigned short* __restrict__ Vb,
                 unsigned short* __restrict__ O,
                 int Bn, int S, int D, int H) {
  __shared__ unsigned short bufK[2][64 * 64]; // [key][hd], double buffered
  __shared__ unsigned short ldsVt[64 * 64];   // transposed: [hd][key]
  __shared__ unsigned short ldsP[4][16 * 64]; // per-wave P scratch [m][key]

  int tid = threadIdx.x, lane = tid & 31, w = tid >> 5;
  int bh = blockIdx.x;
  int b = bh / H, h = bh % H;
  int qblk = blockIdx.y;
  int q0 = qblk * 64 + w * 16;
  size_t headoff = (size_t)h * 64;

  int mA = lane & 15, kb8 = (lane >> 4) * 8;
  int nB = lane & 15, kkb = (lane >> 4) * 8;

  // load Q fragments once (16 rows x 64 k, two k=32 chunks)
  Frag qf[2];
  {
    const unsigned short* qrow = Q + (size_t)(b * S + q0 + mA) * D + headoff;
    qf[0].q[0] = *(const uint4*)&qrow[0 + kb8];
    qf[0].q[1] = *(const uint4*)&qrow[16 + kb8];
    qf[1].q[0] = *(const uint4*)&qrow[32 + kb8];
    qf[1].q[1] = *(const uint4*)&qrow[48 + kb8];
  }

  AccF o[4];
  float mrow[8], lsum[8];
  for (int i = 0; i < 4; i++)
    for (int r = 0; r < 8; r++) o[i].f[r] = 0.f;
  for (int r = 0; r < 8; r++) { mrow[r] = -1e30f; lsum[r] = 0.f; }

  int rstage = tid >> 1, cstage = (tid & 1) * 32;
  int nkb = qblk + 1;   // causal: key blocks up to (and incl.) diagonal
  int cur = 0;

  // prologue: TDM K block 0 -> bufK[0]
  if (w == 0) {
    tdm_load_tile64x64(Kb + (size_t)(b * S) * D + headoff,
                       (unsigned)(uintptr_t)&bufK[0][0], (unsigned)D);
  }

  for (int kbI = 0; kbI < nkb; kbI++) {
    int s0k = kbI * 64;
    __syncthreads();   // prior compute done reading ldsVt / bufK[cur^1]

    // stage V block transposed (overlaps with K DMA)
    {
      const unsigned short* vr =
          Vb + (size_t)(b * S + s0k + rstage) * D + headoff + cstage;
      union { uint4 q[4]; unsigned short s[32]; } t;
      t.q[0] = *(const uint4*)&vr[0];  t.q[1] = *(const uint4*)&vr[8];
      t.q[2] = *(const uint4*)&vr[16]; t.q[3] = *(const uint4*)&vr[24];
#pragma unroll
      for (int i = 0; i < 32; i++) ldsVt[(cstage + i) * 64 + rstage] = t.s[i];
    }

    // wave 0: kick off next K DMA, then wait for the current one
    if (w == 0) {
      if (kbI + 1 < nkb) {
        tdm_load_tile64x64(Kb + (size_t)(b * S + (kbI + 1) * 64) * D + headoff,
                           (unsigned)(uintptr_t)&bufK[cur ^ 1][0], (unsigned)D);
        __builtin_amdgcn_s_wait_tensorcnt(1);   // K(kbI) complete
      } else {
        __builtin_amdgcn_s_wait_tensorcnt(0);
      }
    }
    __syncthreads();

    const unsigned short* ldsK = bufK[cur];

    // ---- scores: 4 tiles of 16 keys, S = Q(16x64) . K^T ----
    AccF st[4];
#pragma unroll
    for (int nt = 0; nt < 4; nt++) {
      int keyrow = nt * 16 + nB;
      Frag k0, k1;
      k0.q[0] = *(const uint4*)&ldsK[keyrow * 64 + kkb];
      k0.q[1] = *(const uint4*)&ldsK[keyrow * 64 + kkb + 16];
      k1.q[0] = *(const uint4*)&ldsK[keyrow * 64 + 32 + kkb];
      k1.q[1] = *(const uint4*)&ldsK[keyrow * 64 + 32 + kkb + 16];
      v8f z = {0.f, 0.f, 0.f, 0.f, 0.f, 0.f, 0.f, 0.f};
      z = wmma_bf16(qf[0], k0, z);
      z = wmma_bf16(qf[1], k1, z);
      st[nt].v = z;
    }

    // ---- scale + causal mask ----
    int mh = (lane >> 4) * 8;
#pragma unroll
    for (int nt = 0; nt < 4; nt++) {
      int keyg = s0k + nt * 16 + nB;
#pragma unroll
      for (int r = 0; r < 8; r++) {
        int mg = q0 + mh + r;
        float s = st[nt].f[r] * 0.125f;   // 1/sqrt(64)
        st[nt].f[r] = (keyg > mg) ? -1e30f : s;
      }
    }

    // ---- online softmax (row lives in one 16-lane half) ----
#pragma unroll
    for (int r = 0; r < 8; r++) {
      float rm = fmaxf(fmaxf(st[0].f[r], st[1].f[r]),
                       fmaxf(st[2].f[r], st[3].f[r]));
      for (int mk = 8; mk > 0; mk >>= 1) rm = fmaxf(rm, __shfl_xor(rm, mk, 32));
      float nm = fmaxf(mrow[r], rm);
      float corr = __expf(mrow[r] - nm);
      mrow[r] = nm;
      float ps = 0.f;
#pragma unroll
      for (int nt = 0; nt < 4; nt++) {
        float p = __expf(st[nt].f[r] - nm);
        st[nt].f[r] = p;
        ps += p;
      }
      for (int mk = 8; mk > 0; mk >>= 1) ps += __shfl_xor(ps, mk, 32);
      lsum[r] = lsum[r] * corr + ps;
#pragma unroll
      for (int ht = 0; ht < 4; ht++) o[ht].f[r] *= corr;
    }

    // ---- C-layout -> A-layout for P via per-wave LDS scratch ----
    unsigned short* Pw = ldsP[w];
#pragma unroll
    for (int nt = 0; nt < 4; nt++)
#pragma unroll
      for (int r = 0; r < 8; r++)
        Pw[(mh + r) * 64 + nt * 16 + nB] = f2b(st[nt].f[r]);

    // ---- O += P(16x64) . V(64x64) ----
#pragma unroll
    for (int kc = 0; kc < 2; kc++) {
      Frag pf;
      pf.q[0] = *(const uint4*)&Pw[mA * 64 + kc * 32 + kb8];
      pf.q[1] = *(const uint4*)&Pw[mA * 64 + kc * 32 + 16 + kb8];
#pragma unroll
      for (int ht = 0; ht < 4; ht++) {
        Frag vf;
        int hd = ht * 16 + nB;
        vf.q[0] = *(const uint4*)&ldsVt[hd * 64 + kc * 32 + kkb];
        vf.q[1] = *(const uint4*)&ldsVt[hd * 64 + kc * 32 + kkb + 16];
        o[ht].v = wmma_bf16(pf, vf, o[ht].v);
      }
    }
    cur ^= 1;
  }

  // ---- epilogue: O /= rowsum, write bf16 ctx ----
  int mh = (lane >> 4) * 8;
  float inv[8];
#pragma unroll
  for (int r = 0; r < 8; r++) inv[r] = 1.0f / lsum[r];
#pragma unroll
  for (int ht = 0; ht < 4; ht++)
#pragma unroll
    for (int r = 0; r < 8; r++) {
      size_t oidx = (size_t)(b * S + q0 + mh + r) * D + headoff + ht * 16 + nB;
      O[oidx] = f2b(o[ht].f[r] * inv[r]);
    }
}

// ---------------------------------------------------------------------------
// Launch
// ---------------------------------------------------------------------------
extern "C" void kernel_launch(void* const* d_in, const int* in_sizes, int n_in,
                              void* d_out, int out_size, void* d_ws, size_t ws_size,
                              hipStream_t stream) {
  const int Bn = 2, S = 2048, D = 1024, H = 16;
  const int M = Bn * S;        // 4096
  const int D4 = 4 * D;        // 4096

  const float* x   = (const float*)d_in[0];
  const float* wq  = (const float*)d_in[1];
  const float* bq  = (const float*)d_in[2];
  const float* wk  = (const float*)d_in[3];
  const float* bk  = (const float*)d_in[4];
  const float* wv  = (const float*)d_in[5];
  const float* bv  = (const float*)d_in[6];
  const float* wo  = (const float*)d_in[7];
  const float* bo  = (const float*)d_in[8];
  const float* w1  = (const float*)d_in[9];
  const float* b1  = (const float*)d_in[10];
  const float* w2  = (const float*)d_in[11];
  const float* b2  = (const float*)d_in[12];
  const float* g1  = (const float*)d_in[13];
  const float* be1 = (const float*)d_in[14];
  const float* g2  = (const float*)d_in[15];
  const float* be2 = (const float*)d_in[16];

  char* ws = (char*)d_ws;
  size_t off = 0;
  auto alloc = [&](size_t bytes) -> void* {
    void* p = ws + off;
    off = (off + bytes + 255) & ~(size_t)255;
    return p;
  };

  unsigned short* hbf  = (unsigned short*)alloc((size_t)M * D * 2);
  unsigned short* wqb  = (unsigned short*)alloc((size_t)D * D * 2);
  unsigned short* wkb  = (unsigned short*)alloc((size_t)D * D * 2);
  unsigned short* wvb  = (unsigned short*)alloc((size_t)D * D * 2);
  unsigned short* wob  = (unsigned short*)alloc((size_t)D * D * 2);
  unsigned short* w1b  = (unsigned short*)alloc((size_t)D * D4 * 2);
  unsigned short* w2b  = (unsigned short*)alloc((size_t)D4 * D * 2);
  unsigned short* qbf  = (unsigned short*)alloc((size_t)M * D * 2);
  unsigned short* kbf  = (unsigned short*)alloc((size_t)M * D * 2);
  unsigned short* vbf  = (unsigned short*)alloc((size_t)M * D * 2);
  unsigned short* ctxb = (unsigned short*)alloc((size_t)M * D * 2);
  float*          x1   = (float*)alloc((size_t)M * D * 4);
  unsigned short* h2b  = (unsigned short*)alloc((size_t)M * D * 2);
  unsigned short* midb = (unsigned short*)alloc((size_t)M * D4 * 2);

  // weight conversions fp32 -> bf16
  cvt_bf16_kernel<<<1024, 256, 0, stream>>>(wq, wqb, D * D);
  cvt_bf16_kernel<<<1024, 256, 0, stream>>>(wk, wkb, D * D);
  cvt_bf16_kernel<<<1024, 256, 0, stream>>>(wv, wvb, D * D);
  cvt_bf16_kernel<<<1024, 256, 0, stream>>>(wo, wob, D * D);
  cvt_bf16_kernel<<<4096, 256, 0, stream>>>(w1, w1b, D * D4);
  cvt_bf16_kernel<<<4096, 256, 0, stream>>>(w2, w2b, D4 * D);

  // LN1 -> bf16 activations
  ln_bf16_kernel<<<M, 256, 0, stream>>>(x, g1, be1, hbf, D);

  // Q/K/V projections (bf16 out)
  gemm_bf16_kernel<<<dim3(D / 128, M / 64), 256, 0, stream>>>(
      hbf, wqb, bq, nullptr, nullptr, qbf, M, D, D, 0);
  gemm_bf16_kernel<<<dim3(D / 128, M / 64), 256, 0, stream>>>(
      hbf, wkb, bk, nullptr, nullptr, kbf, M, D, D, 0);
  gemm_bf16_kernel<<<dim3(D / 128, M / 64), 256, 0, stream>>>(
      hbf, wvb, bv, nullptr, nullptr, vbf, M, D, D, 0);

  // causal flash attention (TDM-staged K)
  attn_kernel<<<dim3(Bn * H, S / 64), 128, 0, stream>>>(
      qbf, kbf, vbf, ctxb, Bn, S, D, H);

  // output projection + residual (fp32 x1)
  gemm_bf16_kernel<<<dim3(D / 128, M / 64), 256, 0, stream>>>(
      ctxb, wob, bo, x, x1, nullptr, M, D, D, 0);

  // LN2 -> bf16
  ln_bf16_kernel<<<M, 256, 0, stream>>>(x1, g2, be2, h2b, D);

  // FFN1 + GELU (bf16 out)
  gemm_bf16_kernel<<<dim3(D4 / 128, M / 64), 256, 0, stream>>>(
      h2b, w1b, b1, nullptr, nullptr, midb, M, D4, D, 1);

  // FFN2 + residual -> final fp32 output
  gemm_bf16_kernel<<<dim3(D / 128, M / 64), 256, 0, stream>>>(
      midb, w2b, b2, x1, (float*)d_out, nullptr, M, D, D4, 0);
}
